// TransformNet_STN2_16234976379275
// MI455X (gfx1250) — matrix operand order repaired and checked by reading, not compiled, not run on hardware
//
#include <hip/hip_runtime.h>
#include <stdint.h>

typedef _Float16 half_t;
typedef __attribute__((ext_vector_type(16))) _Float16 v16h;
typedef __attribute__((ext_vector_type(8)))  _Float16 v8h;
typedef __attribute__((ext_vector_type(8)))  float    v8f;
typedef uint32_t u32x4 __attribute__((ext_vector_type(4)));
typedef int32_t  i32x4 __attribute__((ext_vector_type(4)));
typedef int32_t  i32x8 __attribute__((ext_vector_type(8)));

#define WMMA_F32_F16(a, b, c) \
  __builtin_amdgcn_wmma_f32_16x16x32_f16(false, (a), false, (b), (short)0, (c), false, false)

#if defined(__has_builtin)
#if __has_builtin(__builtin_amdgcn_tensor_load_to_lds) && \
    __has_builtin(__builtin_amdgcn_s_wait_tensorcnt)
#define HAVE_TDM 1
#endif
#endif
#ifndef HAVE_TDM
#define HAVE_TDM 0
#endif

__device__ __forceinline__ int iclamp(int v, int lo, int hi) {
  return v < lo ? lo : (v > hi ? hi : v);
}

#if HAVE_TDM
// 1-D TDM copy: nbytes (multiple of 8) from global gsrc -> LDS lds_dst.
// D# per CDNA5 ISA 08_async_tensor: group0 = {count=1, lds_addr,
// global_addr[56:0], type=2}; group1 = {wg_mask=0, data_size=8B,
// tensor_dim0 = tile_dim0 = n8, tensor_dim1=1, dim0_stride = n8};
// groups 2/3 zero (<=2D tensor). Completion tracked with TENSORcnt.
__device__ __forceinline__ void tdm_load_1d(void* lds_dst, const void* gsrc,
                                            uint32_t nbytes) {
  const uint32_t n8 = nbytes >> 3;
  const uint64_t ga = (uint64_t)(uintptr_t)gsrc;
  const uint32_t la = (uint32_t)(uintptr_t)lds_dst;  // LDS aperture: addr[31:0]
  u32x4 g0;
  g0[0] = 1u;                                          // count=1, user mode
  g0[1] = la;                                          // lds_addr
  g0[2] = (uint32_t)ga;                                // global_addr[31:0]
  g0[3] = (uint32_t)((ga >> 32) & 0x01ffffffu) | (2u << 30);  // [56:32] | type=2
  i32x8 g1;
  g1[0] = (int)(3u << 16);                             // data_size = 8 bytes
  g1[1] = (int)((n8 & 0xffffu) << 16);                 // tensor_dim0[15:0]
  g1[2] = (int)(((n8 >> 16) & 0xffffu) | (1u << 16));  // dim0[31:16], dim1=1
  g1[3] = (int)((n8 & 0xffffu) << 16);                 // tile_dim0 = n8
  g1[4] = 0;                                           // tile_dim1/2 unused
  g1[5] = (int)n8;                                     // dim0_stride[31:0]
  g1[6] = 0;
  g1[7] = 0;
  i32x4 z4 = {0, 0, 0, 0};
#if __has_include(<hip/amd_detail/amd_gfx1250_TDM.h>)
  i32x8 z8 = {0, 0, 0, 0, 0, 0, 0, 0};
  __builtin_amdgcn_tensor_load_to_lds(g0, g1, z4, z4, z8, 0);
#else
  __builtin_amdgcn_tensor_load_to_lds(g0, g1, z4, z4, 0);
#endif
}

__device__ __forceinline__ void tdm_wait() {
  __builtin_amdgcn_s_wait_tensorcnt(0);
  asm volatile("" ::: "memory");  // keep ds_loads below the wait
}
#endif

// A-matrix fragment loader. A is a row-major 16x32 f16 tile pre-packed so
// A[m][k] is the GEMM A element. A 16x32 k order used here (not the ISA table order): lane l ->
// m = l&15, g = l>>4; element e -> K = g*16+e (contiguous, matching B):
// two contiguous 16-byte chunks per lane.
__device__ __forceinline__ v16h load_a_frag(const half_t* __restrict__ A, int lane) {
  const int m = lane & 15, g = lane >> 4;
  v8h lo = *reinterpret_cast<const v8h*>(A + m * 32 + g * 16);
  v8h hi = *reinterpret_cast<const v8h*>(A + m * 32 + g * 16 + 8);
  v16h r;
#pragma unroll
  for (int i = 0; i < 8; ++i) { r[i] = lo[i]; r[i + 8] = hi[i]; }
  return r;
}

// ---------------------------------------------------------------------------
// Kernel 0: pre-pack all weights into WMMA A-tile layouts (f16).
// ---------------------------------------------------------------------------
__global__ __launch_bounds__(256) void prep_weights(
    const float* __restrict__ w1, const float* __restrict__ w2,
    const float* __restrict__ fw1, const float* __restrict__ fw2,
    half_t* __restrict__ Aw1, half_t* __restrict__ Aw2,
    half_t* __restrict__ Afc1, half_t* __restrict__ Afc2) {
  const int tid = threadIdx.x;
  for (int i = tid; i < 9 * 512; i += 256) {
    const int t = i >> 9, r = i & 511, m = r >> 5, k = r & 31;
    Aw1[i] = (half_t)w1[m * 288 + k * 9 + t];  // w1[16][32][3][3]
  }
  for (int i = tid; i < 5 * 512; i += 256) {
    const int s = i >> 9, r = i & 511, m = r >> 5, k = r & 31;
    const int t = 2 * s + (k >> 4), ic = k & 15;
    Aw2[i] = (half_t)((m < 10 && t < 9) ? w2[m * 144 + ic * 9 + t] : 0.f);
  }
  for (int i = tid; i < 6 * 512; i += 256) {
    const int a = i >> 9, r = i & 511, m = r >> 5, k = r & 31;
    const int M = (a >= 3 ? 16 : 0) + m;
    const int K = (a % 3) * 32 + k;
    Afc1[i] = (half_t)((K < 90) ? fw1[M * 90 + K] : 0.f);
  }
  for (int i = tid; i < 512; i += 256) {
    const int m = i >> 5, k = i & 31;
    Afc2[i] = (half_t)((m < 6) ? fw2[m * 32 + k] : 0.f);
  }
}

// ---------------------------------------------------------------------------
// Kernel 1: conv1 (implicit GEMM, M=16 oc, K=288, N=256) + 2x2 maxpool +
// relu. One block per image; x staged into LDS as [pos][32ch] f16.
// Output s1 NHWC f16: [B][8][8][16].
// ---------------------------------------------------------------------------
__global__ __launch_bounds__(256) void conv1_pool(
    const float* __restrict__ x, const half_t* __restrict__ Aw1,
    const float* __restrict__ b1, half_t* __restrict__ s1) {
  __shared__ half_t lx[324 * 32];      // 20.25 KB
  __shared__ float  cb[16 * 16 * 16];  // 16 KB
  const int b = blockIdx.x, tid = threadIdx.x;
  const float* xb = x + (size_t)b * 10368;
  for (int i = tid; i < 10368; i += 256) {   // [c][pos] -> [pos][c] f16
    const int c = i / 324, p = i % 324;
    lx[p * 32 + c] = (half_t)xb[i];
  }
  __syncthreads();

  const int w = tid >> 5, lane = tid & 31;
  const int g = lane >> 4, nx = lane & 15;
  v8f bias;
#pragma unroll
  for (int v = 0; v < 8; ++v) bias[v] = b1[v + 8 * g];

#pragma unroll
  for (int h = 0; h < 2; ++h) {
    const int oy = w + 8 * h;
    v8f acc = bias;
#pragma unroll
    for (int t = 0; t < 9; ++t) {
      const int ky = t / 3, kx = t % 3;
      const v16h a = load_a_frag(Aw1 + t * 512, lane);
      const v16h bm =
          *reinterpret_cast<const v16h*>(&lx[((oy + ky) * 18 + nx + kx) * 32 + g * 16]);
      acc = WMMA_F32_F16(a, bm, acc);
    }
#pragma unroll
    for (int v = 0; v < 8; ++v) cb[(oy * 16 + nx) * 16 + (v + 8 * g)] = acc[v];
  }
  __syncthreads();

  for (int i = tid; i < 1024; i += 256) {
    const int oc = i & 15, px = (i >> 4) & 7, py = i >> 7;
    const float* c00 = &cb[((2 * py) * 16 + 2 * px) * 16 + oc];
    float m = fmaxf(fmaxf(c00[0], c00[16]), fmaxf(c00[256], c00[256 + 16]));
    s1[(size_t)(b * 64 + py * 8 + px) * 16 + oc] = (half_t)fmaxf(m, 0.f);
  }
}

// ---------------------------------------------------------------------------
// Kernel 2: conv2 (M=16(10), K=160(144), N=36/image) + pool + relu.
// One wave per image. The wave's 2 KB s1 tile is DMA'd into LDS by the
// Tensor Data Mover (TENSORcnt), then WMMA B-fragments come from ds_load.
// Output xs[B][96] f16, reference flatten order, cols 90..95 zero.
// ---------------------------------------------------------------------------
__global__ __launch_bounds__(256) void conv2_pool(
    const half_t* __restrict__ s1, const half_t* __restrict__ Aw2,
    const float* __restrict__ b2, half_t* __restrict__ xs) {
  __shared__ half_t ls1[8 * 1024];   // per-wave image tile [y][x][16ch], 16 KB
  __shared__ float  cb[8 * 36 * 16]; // conv2 out [p][oc], 18 KB
  const int tid = threadIdx.x, w = tid >> 5, lane = tid & 31;
  const int g = lane >> 4;
  const int b = blockIdx.x * 8 + w;
  half_t* my = &ls1[w * 1024];

#if HAVE_TDM
  tdm_load_1d(my, s1 + (size_t)b * 1024, 2048);   // whole [8][8][16] f16 tile
#else
  for (int i = lane; i < 256; i += 32) {          // fallback: b64 per lane
    ((uint64_t*)my)[i] = ((const uint64_t*)(s1 + (size_t)b * 1024))[i];
  }
#endif

  v8f bias;
#pragma unroll
  for (int v = 0; v < 8; ++v) {
    const int oc = v + 8 * g;
    bias[v] = (oc < 10) ? b2[oc] : 0.f;
  }

#if HAVE_TDM
  tdm_wait();
#endif

#pragma unroll
  for (int tile = 0; tile < 3; ++tile) {
    const int praw = tile * 16 + (lane & 15);
    const int p = praw > 35 ? 35 : praw;
    const int oy = p / 6, ox = p % 6;
    v8f acc = bias;
#pragma unroll
    for (int s = 0; s < 5; ++s) {
      int t = 2 * s + g;                 // lane group selects the tap
      if (t > 8) t = 8;                  // K>=144 has zero A columns anyway
      const int yy = oy + t / 3, xx = ox + t % 3;
      const v16h a = load_a_frag(Aw2 + s * 512, lane);
      const v16h bm = *reinterpret_cast<const v16h*>(my + (yy * 8 + xx) * 16);
      acc = WMMA_F32_F16(a, bm, acc);
    }
    if (praw < 36) {
#pragma unroll
      for (int v = 0; v < 8; ++v) cb[(w * 36 + p) * 16 + (v + 8 * g)] = acc[v];
    }
  }
  __syncthreads();

  for (int i = tid; i < 768; i += 256) {
    const int img = i / 96, j = i % 96;
    float val = 0.f;
    if (j < 90) {
      const int oc = j / 9, r = j % 9, py = r / 3, px = r % 3;
      const float* c00 = &cb[(img * 36 + (2 * py) * 6 + 2 * px) * 16 + oc];
      val = fmaxf(fmaxf(fmaxf(c00[0], c00[16]), fmaxf(c00[96], c00[96 + 16])), 0.f);
    }
    xs[(size_t)(blockIdx.x * 8 + img) * 96 + j] = (half_t)val;
  }
}

// ---------------------------------------------------------------------------
// Kernel 3: fc1 (relu) + fc2 -> theta[B][6] f32. One wave per 16-batch tile.
// The wave's 3 KB slice of xs is DMA'd to LDS via TDM; h1 is bounced through
// LDS to relayout D->B between the two GEMMs.
// ---------------------------------------------------------------------------
__global__ __launch_bounds__(256) void fc_theta(
    const half_t* __restrict__ xs, const half_t* __restrict__ Afc1,
    const float* __restrict__ fb1, const half_t* __restrict__ Afc2,
    const float* __restrict__ fb2, float* __restrict__ theta) {
  __shared__ half_t lxs[8 * 1536];    // [wave][16 rows][96] f16, 24 KB
  __shared__ half_t h1[8 * 16 * 32];  // 8 KB
  const int tid = threadIdx.x, w = tid >> 5, lane = tid & 31;
  const int g = lane >> 4, nl = lane & 15;
  const int n0 = (blockIdx.x * 8 + w) * 16;
  half_t* my = &lxs[w * 1536];

#if HAVE_TDM
  tdm_load_1d(my, xs + (size_t)n0 * 96, 3072);
#else
  for (int i = lane; i < 384; i += 32) {
    ((uint64_t*)my)[i] = ((const uint64_t*)(xs + (size_t)n0 * 96))[i];
  }
#endif

  v8f acc0, acc1;
#pragma unroll
  for (int v = 0; v < 8; ++v) {
    acc0[v] = fb1[v + 8 * g];
    acc1[v] = fb1[16 + v + 8 * g];
  }

#if HAVE_TDM
  tdm_wait();
#endif

#pragma unroll
  for (int s = 0; s < 3; ++s) {
    const v16h bm = *reinterpret_cast<const v16h*>(my + nl * 96 + s * 32 + g * 16);
    const v16h a0 = load_a_frag(Afc1 + s * 512, lane);
    const v16h a1 = load_a_frag(Afc1 + (3 + s) * 512, lane);
    acc0 = WMMA_F32_F16(a0, bm, acc0);
    acc1 = WMMA_F32_F16(a1, bm, acc1);
  }
#pragma unroll
  for (int v = 0; v < 8; ++v) {
    h1[(w * 16 + nl) * 32 + (v + 8 * g)]      = (half_t)fmaxf(acc0[v], 0.f);
    h1[(w * 16 + nl) * 32 + (16 + v + 8 * g)] = (half_t)fmaxf(acc1[v], 0.f);
  }
  __syncthreads();

  v8f acc2;
#pragma unroll
  for (int v = 0; v < 8; ++v) {
    const int m = v + 8 * g;
    acc2[v] = (m < 6) ? fb2[m] : 0.f;
  }
  const v16h bm2 = *reinterpret_cast<const v16h*>(&h1[(w * 16 + nl) * 32 + g * 16]);
  const v16h a2 = load_a_frag(Afc2, lane);
  acc2 = WMMA_F32_F16(a2, bm2, acc2);
  if (g == 0) {
#pragma unroll
    for (int v = 0; v < 6; ++v) theta[(size_t)(n0 + nl) * 6 + v] = acc2[v];
  }
}

// ---------------------------------------------------------------------------
// Kernel 4: affine_grid + bilinear grid_sample (zeros padding, align=false).
// ---------------------------------------------------------------------------
__global__ __launch_bounds__(256) void stn_sample(
    const float* __restrict__ x, const float* __restrict__ theta,
    float* __restrict__ out) {
  __shared__ int   sx0[324], sy0[324];
  __shared__ float sw00[324], sw10[324], sw01[324], sw11[324];
  const int b = blockIdx.x, tid = threadIdx.x;
  const float t0 = theta[b * 6 + 0], t1 = theta[b * 6 + 1], t2 = theta[b * 6 + 2];
  const float t3 = theta[b * 6 + 3], t4 = theta[b * 6 + 4], t5 = theta[b * 6 + 5];

  for (int pos = tid; pos < 324; pos += 256) {
    const int py = pos / 18, px = pos % 18;
    const float xn = (2.f * px + 1.f) / 18.f - 1.f;
    const float yn = (2.f * py + 1.f) / 18.f - 1.f;
    const float gx = t0 * xn + t1 * yn + t2;
    const float gy = t3 * xn + t4 * yn + t5;
    const float ix = ((gx + 1.f) * 18.f - 1.f) * 0.5f;
    const float iy = ((gy + 1.f) * 18.f - 1.f) * 0.5f;
    const float x0f = floorf(ix), y0f = floorf(iy);
    const float wx1 = ix - x0f, wx0 = 1.f - wx1;
    const float wy1 = iy - y0f, wy0 = 1.f - wy1;
    const int x0 = (int)x0f, y0 = (int)y0f;
    const bool vx0 = (x0 >= 0) & (x0 < 18), vx1 = (x0 + 1 >= 0) & (x0 + 1 < 18);
    const bool vy0 = (y0 >= 0) & (y0 < 18), vy1 = (y0 + 1 >= 0) & (y0 + 1 < 18);
    sx0[pos] = x0; sy0[pos] = y0;
    sw00[pos] = (vx0 & vy0) ? wx0 * wy0 : 0.f;
    sw10[pos] = (vx1 & vy0) ? wx1 * wy0 : 0.f;
    sw01[pos] = (vx0 & vy1) ? wx0 * wy1 : 0.f;
    sw11[pos] = (vx1 & vy1) ? wx1 * wy1 : 0.f;
  }
  __syncthreads();

  const float* xb = x + (size_t)b * 10368;
  float* ob = out + (size_t)b * 10368;
  for (int i = tid; i < 10368; i += 256) {
    const int c = i / 324, pos = i % 324;
    const int x0 = sx0[pos], y0 = sy0[pos];
    const int x0c = iclamp(x0, 0, 17), x1c = iclamp(x0 + 1, 0, 17);
    const int y0c = iclamp(y0, 0, 17), y1c = iclamp(y0 + 1, 0, 17);
    const float* xc = xb + c * 324;
    ob[i] = sw00[pos] * xc[y0c * 18 + x0c] + sw10[pos] * xc[y0c * 18 + x1c] +
            sw01[pos] * xc[y1c * 18 + x0c] + sw11[pos] * xc[y1c * 18 + x1c];
  }
}

// ---------------------------------------------------------------------------
extern "C" void kernel_launch(void* const* d_in, const int* in_sizes, int n_in,
                              void* d_out, int out_size, void* d_ws, size_t ws_size,
                              hipStream_t stream) {
  (void)in_sizes; (void)n_in; (void)out_size; (void)ws_size;
  const float* x   = (const float*)d_in[0];
  const float* w1  = (const float*)d_in[1];
  const float* b1  = (const float*)d_in[2];
  const float* w2  = (const float*)d_in[3];
  const float* b2  = (const float*)d_in[4];
  const float* fw1 = (const float*)d_in[5];
  const float* fb1 = (const float*)d_in[6];
  const float* fw2 = (const float*)d_in[7];
  const float* fb2 = (const float*)d_in[8];
  float* out = (float*)d_out;

  char* ws = (char*)d_ws;
  size_t off = 0;
  auto alloc = [&](size_t bytes) -> void* {
    off = (off + 255) & ~(size_t)255;
    void* p = ws + off;
    off += bytes;
    return p;
  };
  half_t* Aw1  = (half_t*)alloc(9 * 512 * sizeof(half_t));
  half_t* Aw2  = (half_t*)alloc(5 * 512 * sizeof(half_t));
  half_t* Afc1 = (half_t*)alloc(6 * 512 * sizeof(half_t));
  half_t* Afc2 = (half_t*)alloc(512 * sizeof(half_t));
  half_t* s1   = (half_t*)alloc((size_t)4096 * 64 * 16 * sizeof(half_t));  // 8 MB
  half_t* xsb  = (half_t*)alloc((size_t)4096 * 96 * sizeof(half_t));       // 768 KB
  float*  th   = (float*)alloc((size_t)4096 * 6 * sizeof(float));          // 96 KB

  prep_weights<<<1, 256, 0, stream>>>(w1, w2, fw1, fw2, Aw1, Aw2, Afc1, Afc2);
  conv1_pool<<<4096, 256, 0, stream>>>(x, Aw1, b1, s1);
  conv2_pool<<<512, 256, 0, stream>>>(s1, Aw2, b2, xsb);
  fc_theta<<<32, 256, 0, stream>>>(xsb, Afc1, fb1, Afc2, fb2, th);
  stn_sample<<<4096, 256, 0, stream>>>(x, th, out);
}
